// GNNLayer_14207751815182
// MI455X (gfx1250) — compile-verified, matrix operand order checked by reading, not run on hardware
//
#include <hip/hip_runtime.h>
#include <hip/hip_bf16.h>

typedef _Float16 v16h __attribute__((ext_vector_type(16)));
typedef float    v8f  __attribute__((ext_vector_type(8)));

#define IN_DIM  128
#define ATTN    64

// monotonic float -> u32 key (order-preserving), key 0 == "empty/-inf"
__device__ __forceinline__ unsigned fkey(float x) {
    unsigned b = __float_as_uint(x);
    return (b & 0x80000000u) ? ~b : (b | 0x80000000u);
}
__device__ __forceinline__ float fdec(unsigned k) {
    if (k == 0u) return 0.0f;                       // empty segment -> 0 (isneginf rule)
    return (k & 0x80000000u) ? __uint_as_float(k & 0x7fffffffu)
                             : __uint_as_float(~k);
}

// ---------------- init agg keys to 0 ----------------
__global__ __launch_bounds__(256) void init_kernel(uint4* p, int n4) {
    int i = blockIdx.x * 256 + threadIdx.x;
    int stride = gridDim.x * 256;
    uint4 z = {0u, 0u, 0u, 0u};
    for (; i < n4; i += stride) p[i] = z;
}

// ------- tiny table GEMMs: hrWr[r][j] = rela[r]@Wr ; qrW[b][j] = rela[q_rel[b]]@Wqr + bqr -------
__global__ __launch_bounds__(256) void tables_kernel(
    const float* __restrict__ rela, const float* __restrict__ Wr,
    const float* __restrict__ Wqr, const float* __restrict__ bqr,
    const int* __restrict__ q_rel,
    float* __restrict__ hrWr, float* __restrict__ qrW,
    int nRelRows, int batch)
{
    int t = blockIdx.x * 256 + threadIdx.x;
    int nh = nRelRows * ATTN;
    if (t < nh) {
        int r = t >> 6, j = t & 63;
        const float* a = rela + (size_t)r * IN_DIM;
        float acc = 0.f;
        #pragma unroll 8
        for (int k = 0; k < IN_DIM; ++k) acc += a[k] * Wr[k * ATTN + j];
        hrWr[t] = acc;
    } else if (t < nh + batch * ATTN) {
        int t2 = t - nh;
        int b = t2 >> 6, j = t2 & 63;
        const float* a = rela + (size_t)q_rel[b] * IN_DIM;
        float acc = bqr[j];
        #pragma unroll 8
        for (int k = 0; k < IN_DIM; ++k) acc += a[k] * Wqr[k * ATTN + j];
        qrW[t2] = acc;
    }
}

// K index pattern for 16-bit WMMA A/B fragments (16x32 per chunk):
// lane<16: K = {0..7, 16..23}; lane>=16: K = {8..15, 24..31}
__device__ __forceinline__ int kpat(int e, int s) { return s + e + ((e >= 8) ? 8 : 0); }

// ---------------- edge kernel: attn WMMA + alpha + scatter-max ----------------
__global__ __launch_bounds__(256) void edge_kernel(
    const float* __restrict__ hidden, const int* __restrict__ edges,
    const float* __restrict__ rela,   const float* __restrict__ Ws,
    const float* __restrict__ wa,     const float* __restrict__ ba,
    const float* __restrict__ hrWr,   const float* __restrict__ qrW,
    unsigned* __restrict__ aggk, int n_edge)
{
    __shared__ v16h wsFrag[4 * 4 * 32];   // [kcI][nt][lane], 16 KB

    int tid  = threadIdx.x;
    int lane = tid & 31;
    int wave = tid >> 5;

    // pre-swizzle Ws (f32 -> f16) into B-fragment layout
    #pragma unroll
    for (int i = 0; i < 2; ++i) {
        int ent = tid * 2 + i;            // 0..511
        int l   = ent & 31;
        int nt  = (ent >> 5) & 3;
        int kcI = ent >> 7;
        int s   = (l < 16) ? 0 : 8;
        int n   = l & 15;
        v16h b;
        #pragma unroll
        for (int e = 0; e < 16; ++e) {
            int k = kcI * 32 + kpat(e, s);
            b[e] = (_Float16)Ws[k * ATTN + nt * 16 + n];
        }
        wsFrag[ent] = b;
    }
    __syncthreads();

    int tile = blockIdx.x * 8 + wave;
    int e0   = tile * 16;
    if (e0 >= n_edge) return;

    int m  = lane & 15;
    int em = e0 + m;
    if (em >= n_edge) em = n_edge - 1;    // duplicate tail edge: idempotent under max
    const int* ep = edges + (size_t)em * 6;
    int ridxv = ep[0];
    int relv  = ep[2];
    int subv  = ep[4];
    int objv  = ep[5];

    int s  = (lane < 16) ? 0 : 8;
    v8f z = {};
    v8f c0 = z, c1 = z, c2 = z, c3 = z;

    // attn = hs @ Ws  (K = 128 in 4 chunks of 32)
    #pragma unroll
    for (int kcI = 0; kcI < 4; ++kcI) {
        const float4* hp = (const float4*)(hidden + (size_t)subv * IN_DIM + kcI * 32 + s);
        float4 f0 = hp[0], f1 = hp[1], f2 = hp[4], f3 = hp[5];
        v16h a;
        a[0]=(_Float16)f0.x; a[1]=(_Float16)f0.y; a[2]=(_Float16)f0.z; a[3]=(_Float16)f0.w;
        a[4]=(_Float16)f1.x; a[5]=(_Float16)f1.y; a[6]=(_Float16)f1.z; a[7]=(_Float16)f1.w;
        a[8]=(_Float16)f2.x; a[9]=(_Float16)f2.y; a[10]=(_Float16)f2.z; a[11]=(_Float16)f2.w;
        a[12]=(_Float16)f3.x; a[13]=(_Float16)f3.y; a[14]=(_Float16)f3.z; a[15]=(_Float16)f3.w;
        c0 = __builtin_amdgcn_wmma_f32_16x16x32_f16(false, a, false, wsFrag[kcI*128 +  0 + lane], (short)0, c0, false, false);
        c1 = __builtin_amdgcn_wmma_f32_16x16x32_f16(false, a, false, wsFrag[kcI*128 + 32 + lane], (short)0, c1, false, false);
        c2 = __builtin_amdgcn_wmma_f32_16x16x32_f16(false, a, false, wsFrag[kcI*128 + 64 + lane], (short)0, c2, false, false);
        c3 = __builtin_amdgcn_wmma_f32_16x16x32_f16(false, a, false, wsFrag[kcI*128 + 96 + lane], (short)0, c3, false, false);
    }

    // add rel / query tables, relu, dot with wa (accumulate per-row partials)
    int mb = (lane < 16) ? 0 : 8;
    int n  = lane & 15;
    float p[8];
    #pragma unroll
    for (int v = 0; v < 8; ++v) p[v] = 0.f;

    v8f cc[4] = {c0, c1, c2, c3};
    #pragma unroll
    for (int nt = 0; nt < 4; ++nt) {
        float wan = wa[nt * 16 + n];
        #pragma unroll
        for (int v = 0; v < 8; ++v) {
            int re = __shfl(relv,  mb + v, 32);
            int ri = __shfl(ridxv, mb + v, 32);
            float t = cc[nt][v] + hrWr[re * ATTN + nt * 16 + n] + qrW[ri * ATTN + nt * 16 + n];
            t = fmaxf(t, 0.f);
            p[v] += t * wan;
        }
    }
    // reduce across 16 lanes sharing the same row half
    #pragma unroll
    for (int v = 0; v < 8; ++v) {
        #pragma unroll
        for (int off = 8; off >= 1; off >>= 1) p[v] += __shfl_xor(p[v], off, 32);
    }
    float bav = ba[0];
    float al[8];
    #pragma unroll
    for (int v = 0; v < 8; ++v) al[v] = 1.0f / (1.0f + __expf(-(p[v] + bav)));

    // scatter: message = alpha*(hs+hr), atomic max into agg keys (lane handles 4 cols)
    #pragma unroll
    for (int mm = 0; mm < 16; ++mm) {
        float alm = (mm < 8) ? __shfl(al[mm], 0, 32) : __shfl(al[mm - 8], 16, 32);
        int su = __shfl(subv, mm, 32);
        int re = __shfl(relv, mm, 32);
        int ob = __shfl(objv, mm, 32);
        float4 h = *(const float4*)(hidden + (size_t)su * IN_DIM + lane * 4);
        float4 r = *(const float4*)(rela   + (size_t)re * IN_DIM + lane * 4);
        unsigned* d = aggk + (size_t)ob * IN_DIM + lane * 4;
        atomicMax(d + 0, fkey(alm * (h.x + r.x)));
        atomicMax(d + 1, fkey(alm * (h.y + r.y)));
        atomicMax(d + 2, fkey(alm * (h.z + r.z)));
        atomicMax(d + 3, fkey(alm * (h.w + r.w)));
    }
}

// ---------------- output GEMM: out = decode(agg) @ W_h ----------------
__global__ __launch_bounds__(256) void out_kernel(
    const unsigned* __restrict__ aggk, const float* __restrict__ W_h,
    float* __restrict__ out, int n_node)
{
    __shared__ v16h whFrag[4 * 8 * 32];   // [kcI][nt][lane], 32 KB

    int tid  = threadIdx.x;
    int lane = tid & 31;
    int wave = tid >> 5;

    #pragma unroll
    for (int i = 0; i < 4; ++i) {
        int ent = tid * 4 + i;            // 0..1023
        int l   = ent & 31;
        int nt  = (ent >> 5) & 7;
        int kcI = ent >> 8;
        int s   = (l < 16) ? 0 : 8;
        int n   = l & 15;
        v16h b;
        #pragma unroll
        for (int e = 0; e < 16; ++e) {
            int k = kcI * 32 + kpat(e, s);
            b[e] = (_Float16)W_h[k * IN_DIM + nt * 16 + n];
        }
        whFrag[ent] = b;
    }
    __syncthreads();

    int mt = blockIdx.x * 8 + wave;
    int m0 = mt * 16;
    if (m0 >= n_node) return;

    int m   = lane & 15;
    int row = m0 + m; if (row >= n_node) row = n_node - 1;
    int s   = (lane < 16) ? 0 : 8;

    v8f z = {};
    v8f c[8];
    #pragma unroll
    for (int nt = 0; nt < 8; ++nt) c[nt] = z;

    #pragma unroll
    for (int kcI = 0; kcI < 4; ++kcI) {
        const uint4* ap = (const uint4*)(aggk + (size_t)row * IN_DIM + kcI * 32 + s);
        uint4 u0 = ap[0], u1 = ap[1], u2 = ap[4], u3 = ap[5];
        v16h a;
        a[0]=(_Float16)fdec(u0.x); a[1]=(_Float16)fdec(u0.y); a[2]=(_Float16)fdec(u0.z); a[3]=(_Float16)fdec(u0.w);
        a[4]=(_Float16)fdec(u1.x); a[5]=(_Float16)fdec(u1.y); a[6]=(_Float16)fdec(u1.z); a[7]=(_Float16)fdec(u1.w);
        a[8]=(_Float16)fdec(u2.x); a[9]=(_Float16)fdec(u2.y); a[10]=(_Float16)fdec(u2.z); a[11]=(_Float16)fdec(u2.w);
        a[12]=(_Float16)fdec(u3.x); a[13]=(_Float16)fdec(u3.y); a[14]=(_Float16)fdec(u3.z); a[15]=(_Float16)fdec(u3.w);
        #pragma unroll
        for (int nt = 0; nt < 8; ++nt)
            c[nt] = __builtin_amdgcn_wmma_f32_16x16x32_f16(false, a, false,
                        whFrag[kcI * 256 + nt * 32 + lane], (short)0, c[nt], false, false);
    }

    int mb = (lane < 16) ? 0 : 8;
    int n  = lane & 15;
    #pragma unroll
    for (int nt = 0; nt < 8; ++nt) {
        #pragma unroll
        for (int v = 0; v < 8; ++v) {
            int r = m0 + mb + v;
            if (r < n_node) out[(size_t)r * IN_DIM + nt * 16 + n] = c[nt][v];
        }
    }
}

extern "C" void kernel_launch(void* const* d_in, const int* in_sizes, int n_in,
                              void* d_out, int out_size, void* d_ws, size_t ws_size,
                              hipStream_t stream) {
    const int*   q_rel  = (const int*)  d_in[1];
    const float* hidden = (const float*)d_in[2];
    const int*   edges  = (const int*)  d_in[3];
    const float* rela   = (const float*)d_in[7];
    const float* Ws     = (const float*)d_in[8];
    const float* Wr     = (const float*)d_in[9];
    const float* Wqr    = (const float*)d_in[10];
    const float* bqr    = (const float*)d_in[11];
    const float* wa     = (const float*)d_in[12];
    const float* ba     = (const float*)d_in[13];
    const float* W_h    = (const float*)d_in[14];

    int n_edge   = in_sizes[3] / 6;
    int n_node   = in_sizes[2] / IN_DIM;
    int nRelRows = in_sizes[7] / IN_DIM;
    int batch    = in_sizes[1];

    // workspace layout: [agg keys: n_node*128 u32][hrWr: nRelRows*64 f32][qrW: batch*64 f32]
    unsigned* aggk = (unsigned*)d_ws;
    float*    hrWr = (float*)((char*)d_ws + (size_t)n_node * IN_DIM * sizeof(unsigned));
    float*    qrW  = hrWr + (size_t)nRelRows * ATTN;

    // 1) zero agg keys
    int n4 = (n_node * IN_DIM) / 4;
    int ib = (n4 + 255) / 256; if (ib > 4096) ib = 4096;
    init_kernel<<<ib, 256, 0, stream>>>((uint4*)aggk, n4);

    // 2) tables
    int ttot = (nRelRows + batch) * ATTN;
    tables_kernel<<<(ttot + 255) / 256, 256, 0, stream>>>(rela, Wr, Wqr, bqr, q_rel,
                                                          hrWr, qrW, nRelRows, batch);

    // 3) per-edge attention + scatter-max (1 wave per 16 edges, 8 waves per block)
    int tiles  = (n_edge + 15) / 16;
    int eblk   = (tiles + 7) / 8;
    edge_kernel<<<eblk, 256, 0, stream>>>(hidden, edges, rela, Ws, wa, ba,
                                          hrWr, qrW, aggk, n_edge);

    // 4) out = decode(agg) @ W_h
    int mtiles = (n_node + 15) / 16;
    int oblk   = (mtiles + 7) / 8;
    out_kernel<<<oblk, 256, 0, stream>>>(aggk, W_h, (float*)d_out, n_node);
}
